// SCCA_IPAdapterAttnProcessor_764504178926
// MI455X (gfx1250) — matrix operand order repaired and checked by reading, not compiled, not run on hardware
//
#include <hip/hip_runtime.h>

// ---------------------------------------------------------------------------
// SCCA IP-Adapter attention for MI455X (gfx1250): bf16 WMMA + async-LDS.
// ---------------------------------------------------------------------------

typedef __attribute__((ext_vector_type(16))) __bf16 v16bf;
typedef __attribute__((ext_vector_type(8)))  float  v8f;

#define NBATCH  4
#define S_LEN   4096
#define D_MODEL 1280
#define CDIM    2048
#define NHEAD   20
#define HEADD   64
#define NKV     77

// ---------------------------------------------------------------------------
// CDNA5 async global->LDS copy (GLOBAL_LOAD_ASYNC_TO_LDS_B128, ASYNCcnt).
// LDS byte address = low 32 bits of the generic pointer (ISA 10.2 aperture).
// ---------------------------------------------------------------------------
__device__ __forceinline__ void async_copy16(const void* g, void* l) {
  unsigned lds = (unsigned)(unsigned long long)(uintptr_t)l;
  unsigned long long ga = (unsigned long long)(uintptr_t)g;
  asm volatile("global_load_async_to_lds_b128 %0, %1, off"
               :: "v"(lds), "v"(ga) : "memory");
}
__device__ __forceinline__ void wait_async0() {
  asm volatile("s_wait_asynccnt 0x0" ::: "memory");
}

// ---------------------------------------------------------------------------
// WMMA fragment loaders (same contiguous k = 16*(lane/16) + i order for A and B, wave32)
// ---------------------------------------------------------------------------

// A-matrix 16x32 bf16 fragment from LDS, row-major with row stride ldrow.
__device__ __forceinline__ v16bf load_frag_a(const __bf16* p, int ldrow) {
  int lane = threadIdx.x & 31;
  int row  = lane & 15;
  int kb   = (lane >> 4) << 4;   // 0 or 16
  const __bf16* q = p + row * ldrow;
  v16bf a;
#pragma unroll
  for (int i = 0; i < 16; ++i) a[i] = q[kb + i];
  return a;
}

// B-matrix 32x16 bf16 fragment; logical B[k][n] stored row-major (stride ldrow).
__device__ __forceinline__ v16bf load_frag_b(const __bf16* p, int ldrow) {
  int lane = threadIdx.x & 31;
  int col  = lane & 15;
  int kb   = (lane >> 4) << 4;   // 0 or 16
  v16bf b;
#pragma unroll
  for (int i = 0; i < 16; ++i) b[i] = p[(kb + i) * ldrow + col];
  return b;
}

// Logical B[k][n] with the LDS tile stored transposed: (k,n) at p[n*ldn + k].
__device__ __forceinline__ v16bf load_frag_b_t(const __bf16* p, int ldn) {
  int lane = threadIdx.x & 31;
  int col  = lane & 15;
  int kb   = (lane >> 4) << 4;
  v16bf b;
#pragma unroll
  for (int i = 0; i < 16; ++i) b[i] = p[col * ldn + kb + i];
  return b;
}

// ---------------------------------------------------------------------------
// Kernel 1: mask ratio -> boost scalar
// ---------------------------------------------------------------------------
__global__ void mask_stats_kernel(const float* __restrict__ mask,
                                  float* __restrict__ boost_out) {
  __shared__ float red[256];
  float s = 0.f;
  for (int i = threadIdx.x; i < S_LEN; i += 256) s += mask[i];
  red[threadIdx.x] = s;
  __syncthreads();
  for (int k = 128; k > 0; k >>= 1) {
    if ((int)threadIdx.x < k) red[threadIdx.x] += red[threadIdx.x + k];
    __syncthreads();
  }
  if (threadIdx.x == 0) {
    float ratio = red[0] * (1.0f / (float)S_LEN);
    float b = ((ratio > 1e-6f) && (ratio < 0.1f))
                  ? fminf(1.0f + (0.1f - ratio) * 30.0f, 4.0f)
                  : 1.0f;
    boost_out[0] = b;
  }
}

// ---------------------------------------------------------------------------
// WMMA GEMM: C[M,N] = A[M,K] @ B[K,N] (+bias), bf16 math, f32 accum.
// Block tile 128x128x32, 256 threads = 8 waves; wave tile 64x32 -> 8 WMMA/step.
// Double-buffered LDS; fp32 operands staged through registers (float4) with
// bf16 conversion; bf16 A operand copied with async global->LDS (b128).
// N, K multiples of 128/32; M bounds-checked.
// ---------------------------------------------------------------------------
template <typename TA, typename TC, bool BIAS>
__global__ __launch_bounds__(256)
void gemm_bf16_wmma(const TA* __restrict__ A,
                    const float* __restrict__ B,
                    TC* __restrict__ C,
                    const float* __restrict__ bias,
                    int M, int N, int K) {
  constexpr int BM = 128, BN = 128, BK = 32;
  constexpr bool A_BF16 = (sizeof(TA) == 2);

  __shared__ __bf16 As[2][BM * BK];
  __shared__ __bf16 Bs[2][BK * BN];

  const int tid  = threadIdx.x;
  const int gm   = blockIdx.y * BM;
  const int gn   = blockIdx.x * BN;
  const int wid  = tid >> 5;
  const int lane = tid & 31;
  const int wm   = (wid >> 2) * 64;   // wave M offset (0 / 64)
  const int wn   = (wid & 3) * 32;    // wave N offset (0/32/64/96)

  v8f acc[4][2] = {};
  float4 ra[4];
  float4 rbv[4];

  // stage: issue global loads for tile at kt (A-async writes dbuf directly)
  auto stage = [&](int kt, int dbuf) {
    if constexpr (A_BF16) {
#pragma unroll
      for (int i = 0; i < 2; ++i) {
        int idx = tid + i * 256;          // 512 x 16B chunks
        int r = idx >> 2, c = idx & 3;
        int ar = gm + r;
        if (ar < M) {
          async_copy16((const char*)(A + (size_t)ar * K + kt) + c * 16,
                       &As[dbuf][r * BK + c * 8]);
        } else {
#pragma unroll
          for (int j = 0; j < 8; ++j) As[dbuf][r * BK + c * 8 + j] = (__bf16)0.0f;
        }
      }
    } else {
#pragma unroll
      for (int i = 0; i < 4; ++i) {
        int idx = tid + i * 256;          // 1024 x float4
        int r = idx >> 3, c4 = idx & 7;
        int ar = gm + r;
        ra[i] = (ar < M) ? *(const float4*)(A + (size_t)ar * K + kt + c4 * 4)
                         : make_float4(0.f, 0.f, 0.f, 0.f);
      }
    }
#pragma unroll
    for (int i = 0; i < 4; ++i) {
      int idx = tid + i * 256;            // 1024 x float4
      int r = idx >> 5, c4 = idx & 31;
      rbv[i] = *(const float4*)(B + (size_t)(kt + r) * N + gn + c4 * 4);
    }
  };

  // commit: convert staged registers to bf16 and store into LDS buffer dbuf
  auto commit = [&](int dbuf) {
    if constexpr (!A_BF16) {
#pragma unroll
      for (int i = 0; i < 4; ++i) {
        int idx = tid + i * 256;
        int r = idx >> 3, c4 = idx & 7;
        const float* f = (const float*)&ra[i];
#pragma unroll
        for (int j = 0; j < 4; ++j) As[dbuf][r * BK + c4 * 4 + j] = (__bf16)f[j];
      }
    }
#pragma unroll
    for (int i = 0; i < 4; ++i) {
      int idx = tid + i * 256;
      int r = idx >> 5, c4 = idx & 31;
      const float* f = (const float*)&rbv[i];
#pragma unroll
      for (int j = 0; j < 4; ++j) Bs[dbuf][r * BN + c4 * 4 + j] = (__bf16)f[j];
    }
  };

  stage(0, 0);
  commit(0);
  if constexpr (A_BF16) wait_async0();
  __syncthreads();

  const int nk = K / BK;
  for (int t = 0; t < nk; ++t) {
    const int buf = t & 1;
    const bool more = (t + 1) < nk;
    if (more) stage((t + 1) * BK, buf ^ 1);
    if (t + 2 < nk) {  // speculative prefetch of the t+2 tiles (global_prefetch_b8)
      __builtin_prefetch(A + (size_t)(gm + (tid >> 1)) * K + (t + 2) * BK, 0, 1);
      __builtin_prefetch(B + (size_t)((t + 2) * BK + (tid & 31)) * N + gn, 0, 1);
    }

    v16bf af[4], bw[2];
#pragma unroll
    for (int i = 0; i < 4; ++i) af[i] = load_frag_a(&As[buf][(wm + i * 16) * BK], BK);
#pragma unroll
    for (int j = 0; j < 2; ++j) bw[j] = load_frag_b(&Bs[buf][wn + j * 16], BN);
#pragma unroll
    for (int i = 0; i < 4; ++i)
#pragma unroll
      for (int j = 0; j < 2; ++j)
        acc[i][j] = __builtin_amdgcn_wmma_f32_16x16x32_bf16(
            false, af[i], false, bw[j], (short)0, acc[i][j], false, false);

    if (more) commit(buf ^ 1);
    if constexpr (A_BF16) wait_async0();
    __syncthreads();
  }

  const int rb  = (lane >> 4) * 8;
  const int col = lane & 15;
#pragma unroll
  for (int i = 0; i < 4; ++i) {
#pragma unroll
    for (int j = 0; j < 2; ++j) {
#pragma unroll
      for (int r = 0; r < 8; ++r) {
        int grow = gm + wm + i * 16 + rb + r;
        if (grow < M) {
          int gc = gn + wn + j * 16 + col;
          float v = acc[i][j][r];
          if (BIAS) v += bias[gc];
          C[(size_t)grow * N + gc] = (TC)v;
        }
      }
    }
  }
}

// ---------------------------------------------------------------------------
// Fused dual-branch attention + mask/boost combine.
// Grid: (S/64, B*H).  128 threads = 4 waves, each wave owns 16 query rows.
// Q/K/V tiles fetched with async global->LDS b128 copies.
// ---------------------------------------------------------------------------
__global__ __launch_bounds__(128)
void attn_fused_kernel(const __bf16* __restrict__ Qb,   // [B*S, D]
                       const __bf16* __restrict__ Kb,   // [B*77, D]
                       const __bf16* __restrict__ Vb,
                       const __bf16* __restrict__ Kib,
                       const __bf16* __restrict__ Vib,
                       const float*  __restrict__ maskp, // [4096]
                       const float*  __restrict__ boostp,
                       __bf16* __restrict__ Ob) {        // [B*S, D]
  __shared__ __align__(16) unsigned char smem[53504];
  __bf16* qs   = (__bf16*)(smem);            // [64][64]
  __bf16* ks   = (__bf16*)(smem + 8192);     // [96][64] zero-padded keys
  __bf16* vs   = (__bf16*)(smem + 20480);    // [96][64]
  float*  sc   = (float*)(smem + 32768);     // [4][16*80] (time-sliced with pb)
  __bf16* pb   = (__bf16*)(smem + 32768);    // [4][16*96]
  float*  sums = (float*)(smem + 53248);     // [64]

  const int tid  = threadIdx.x;
  const int lane = tid & 31;
  const int wid  = tid >> 5;           // 0..3
  const int q0   = blockIdx.x * 64;    // query tile start within S
  const int b    = blockIdx.y / NHEAD;
  const int h    = blockIdx.y % NHEAD;
  const float boost = boostp[0];

  // ---- Q tile (64 rows x 64 dims): async copy, 512 x 16B chunks ----
  for (int i = tid; i < 64 * 8; i += 128) {
    int r = i >> 3, c = i & 7;
    async_copy16(Qb + (size_t)(b * S_LEN + q0 + r) * D_MODEL + h * HEADD + c * 8,
                 qs + r * 64 + c * 8);
  }
  wait_async0();
  __syncthreads();

  v16bf aq0 = load_frag_a(qs + (wid * 16) * 64, 64);
  v16bf aq1 = load_frag_a(qs + (wid * 16) * 64 + 32, 64);

  const int rb  = (lane >> 4) * 8;
  const int col = lane & 15;
  float out0[32];

#pragma unroll 1
  for (int br = 0; br < 2; ++br) {
    const __bf16* Kp = br ? Kib : Kb;
    const __bf16* Vp = br ? Vib : Vb;

    __syncthreads();  // previous branch done reading ks/vs
    for (int i = tid; i < 96 * 8; i += 128) {   // 768 x 16B chunks per matrix
      int r = i >> 3, c = i & 7;
      if (r < NKV) {
        size_t g = (size_t)(b * NKV + r) * D_MODEL + h * HEADD + c * 8;
        async_copy16(Kp + g, ks + r * 64 + c * 8);
        async_copy16(Vp + g, vs + r * 64 + c * 8);
      } else {
#pragma unroll
        for (int j = 0; j < 8; ++j) {
          ks[r * 64 + c * 8 + j] = (__bf16)0.0f;
          vs[r * 64 + c * 8 + j] = (__bf16)0.0f;
        }
      }
    }
    wait_async0();
    __syncthreads();

    // ---- scores: S[16,80] = Q[16,64] x K^T[64,80] ----
    float* scw = sc + wid * (16 * 80);
#pragma unroll
    for (int nt = 0; nt < 5; ++nt) {
      v8f accs = {};
      v16bf bk0 = load_frag_b_t(ks + (nt * 16) * 64, 64);
      v16bf bk1 = load_frag_b_t(ks + (nt * 16) * 64 + 32, 64);
      accs = __builtin_amdgcn_wmma_f32_16x16x32_bf16(false, aq0, false, bk0, (short)0, accs, false, false);
      accs = __builtin_amdgcn_wmma_f32_16x16x32_bf16(false, aq1, false, bk1, (short)0, accs, false, false);
#pragma unroll
      for (int r = 0; r < 8; ++r)
        scw[(rb + r) * 80 + nt * 16 + col] = accs[r];
    }
    __syncthreads();

    // ---- softmax: 2 lanes per row, 40 keys each ----
    {
      const int row  = lane & 15;
      const int half = lane >> 4;
      const float* srow = sc + wid * (16 * 80) + row * 80 + half * 40;
      float buf[40];
      float mx = -1e30f;
#pragma unroll
      for (int j = 0; j < 40; ++j) {
        int n = half * 40 + j;
        float x = (n < NKV) ? srow[j] * 0.125f : -1e30f;  // 1/sqrt(64)
        buf[j] = x;
        mx = fmaxf(mx, x);
      }
      mx = fmaxf(mx, __shfl_xor(mx, 16, 32));
      float sum = 0.f;
#pragma unroll
      for (int j = 0; j < 40; ++j) {
        float e = __expf(buf[j] - mx);
        buf[j] = e;
        sum += e;
      }
      sum += __shfl_xor(sum, 16, 32);
      __syncthreads();  // all waves done reading scores; pb overlay safe

      __bf16* prow = pb + wid * (16 * 96) + row * 96;
#pragma unroll
      for (int j = 0; j < 40; ++j) prow[half * 40 + j] = (__bf16)buf[j];
#pragma unroll
      for (int j = 0; j < 8; ++j) prow[80 + half * 8 + j] = (__bf16)0.0f;
      if (half == 0) sums[wid * 16 + row] = sum;
    }
    __syncthreads();

    // ---- PV: O[16,64] = P[16,96] x V[96,64] ----
    v8f acco[4];
#pragma unroll
    for (int nt = 0; nt < 4; ++nt) acco[nt] = (v8f){};
#pragma unroll
    for (int k0 = 0; k0 < 96; k0 += 32) {
      v16bf ap = load_frag_a(pb + wid * (16 * 96) + k0, 96);
#pragma unroll
      for (int nt = 0; nt < 4; ++nt) {
        v16bf bv = load_frag_b(vs + k0 * 64 + nt * 16, 64);
        acco[nt] = __builtin_amdgcn_wmma_f32_16x16x32_bf16(false, ap, false, bv, (short)0, acco[nt], false, false);
      }
    }

    // ---- normalize; branch 0 stashed, branch 1 combined + stored ----
#pragma unroll
    for (int nt = 0; nt < 4; ++nt) {
#pragma unroll
      for (int r = 0; r < 8; ++r) {
        int m = rb + r;
        float inv = 1.0f / sums[wid * 16 + m];
        float v = acco[nt][r] * inv;
        if (br == 0) {
          out0[nt * 8 + r] = v;
        } else {
          int qrow = q0 + wid * 16 + m;                 // 0..4095
          float res = out0[nt * 8 + r] + boost * maskp[qrow] * v;
          Ob[(size_t)(b * S_LEN + qrow) * D_MODEL + h * HEADD + nt * 16 + col] = (__bf16)res;
        }
      }
    }
  }
}

// ---------------------------------------------------------------------------
// Launcher
// ---------------------------------------------------------------------------
extern "C" void kernel_launch(void* const* d_in, const int* in_sizes, int n_in,
                              void* d_out, int out_size, void* d_ws, size_t ws_size,
                              hipStream_t stream) {
  const float* hs   = (const float*)d_in[0];   // [4,4096,1280]
  const float* ehs  = (const float*)d_in[1];   // [4,77,2048]
  const float* mask = (const float*)d_in[2];   // [1,1,64,64] -> flat 4096
  const float* Wq   = (const float*)d_in[3];
  const float* Wk   = (const float*)d_in[4];
  const float* Wv   = (const float*)d_in[5];
  const float* Wkip = (const float*)d_in[6];
  const float* Wvip = (const float*)d_in[7];
  const float* Wo   = (const float*)d_in[8];
  const float* bo   = (const float*)d_in[9];
  float* out = (float*)d_out;

  const int MQ  = NBATCH * S_LEN;   // 16384
  const int MKV = NBATCH * NKV;     // 308

  char* w = (char*)d_ws;
  __bf16* qb  = (__bf16*)w;  w += (size_t)MQ * D_MODEL * 2;
  __bf16* kb  = (__bf16*)w;  w += (size_t)MKV * D_MODEL * 2;
  __bf16* vb  = (__bf16*)w;  w += (size_t)MKV * D_MODEL * 2;
  __bf16* kib = (__bf16*)w;  w += (size_t)MKV * D_MODEL * 2;
  __bf16* vib = (__bf16*)w;  w += (size_t)MKV * D_MODEL * 2;
  __bf16* ab  = (__bf16*)w;  w += (size_t)MQ * D_MODEL * 2;
  float*  boost = (float*)w;

  // 1) mask statistics -> boost
  mask_stats_kernel<<<1, 256, 0, stream>>>(mask, boost);

  // 2) Q projection: [16384,1280] = hs @ Wq
  dim3 gq(D_MODEL / 128, MQ / 128);     // 10 x 128
  gemm_bf16_wmma<float, __bf16, false><<<gq, 256, 0, stream>>>(hs, Wq, qb, nullptr, MQ, D_MODEL, D_MODEL);

  // 3) K/V projections (4x): [308,1280] = ehs @ W*
  dim3 gkv(D_MODEL / 128, (MKV + 127) / 128);  // 10 x 3
  gemm_bf16_wmma<float, __bf16, false><<<gkv, 256, 0, stream>>>(ehs, Wk,   kb,  nullptr, MKV, D_MODEL, CDIM);
  gemm_bf16_wmma<float, __bf16, false><<<gkv, 256, 0, stream>>>(ehs, Wv,   vb,  nullptr, MKV, D_MODEL, CDIM);
  gemm_bf16_wmma<float, __bf16, false><<<gkv, 256, 0, stream>>>(ehs, Wkip, kib, nullptr, MKV, D_MODEL, CDIM);
  gemm_bf16_wmma<float, __bf16, false><<<gkv, 256, 0, stream>>>(ehs, Wvip, vib, nullptr, MKV, D_MODEL, CDIM);

  // 4) fused dual-branch attention + mask/boost combine -> bf16 activations
  attn_fused_kernel<<<dim3(S_LEN / 64, NBATCH * NHEAD), 128, 0, stream>>>(
      qb, kb, vb, kib, vib, mask, boost, ab);

  // 5) output projection with bias -> f32 d_out
  gemm_bf16_wmma<__bf16, float, true><<<gq, 256, 0, stream>>>(ab, Wo, out, bo, MQ, D_MODEL, D_MODEL);
}
